// SCDMNSliced_19284403159049
// MI455X (gfx1250) — matrix operand order repaired and checked by reading, not compiled, not run on hardware
//
#include <hip/hip_runtime.h>
#include <hip/hip_bf16.h>

typedef __attribute__((ext_vector_type(16))) _Float16 v16h;
typedef __attribute__((ext_vector_type(8)))  float    v8f;
typedef __attribute__((ext_vector_type(8)))  _Float16 h8;   // 16 bytes

#define BN_EPS 1e-5f

#if __has_builtin(__builtin_amdgcn_global_load_async_to_lds_b128) && \
    __has_builtin(__builtin_amdgcn_s_wait_asynccnt)
#define USE_ASYNC_LDS 1
typedef int v4ib __attribute__((vector_size(16)));            // gcc-style int4
typedef __attribute__((address_space(1))) v4ib* GAS_P;        // global AS
typedef __attribute__((address_space(3))) v4ib* LAS_P;        // LDS AS
#define ASYNC_CP16(gp_, lp_) \
  __builtin_amdgcn_global_load_async_to_lds_b128((GAS_P)(gp_), (LAS_P)(lp_), 0, 0)
#else
#define USE_ASYNC_LDS 0
#endif

// ---------------------------------------------------------------------------
// Fast implicit-GEMM conv, NHWC f16 activations, pre-transformed f16 weights
// [co][rs][ci] so GEMM-K (k = rs*Cin + ci) is memory-contiguous.
// Workgroup = 128 threads = 4 wave32s. Tile: 128(M=pixels) x 64(N=Cout),
// K-step 32, DOUBLE-BUFFERED LDS: next K-tile is staged (async global->LDS
// B128 when available, else reg-staged b128 copies) while the current tile's
// 8 v_wmma_f32_16x16x32_f16 per wave execute. One barrier per K-step.
// Cin, HoWo are powers of two -> all index math is shifts.
// EPI: 1 = BN+ReLU*mask -> f16   (conv1)
//      2 = BN -> f16             (projection shortcut)
//      3 = (BN+sc)*mask, ReLU -> f32 (conv2 / block output)
// ---------------------------------------------------------------------------
template<int EPI, int RR, int SS>
__global__ __launch_bounds__(128) void conv_wmma_nhwc(
    const _Float16* __restrict__ in, const _Float16* __restrict__ w16,
    const float* __restrict__ bng, const float* __restrict__ bnb,
    const float* __restrict__ bnrm, const float* __restrict__ bnrv,
    const float* __restrict__ mask, const _Float16* __restrict__ sc,
    void* __restrict__ outp,
    int cinShift, int Hin, int Win, int Cout, int Wo, int hwShift, int stride)
{
  constexpr int RS  = RR * SS;
  constexpr int PAD = (RR == 3) ? 1 : 0;
  const int Cin = 1 << cinShift;
  const int K   = Cin * RS;            // always a multiple of 32 here (>= 64)

  __shared__ alignas(32) _Float16 Ash[2][8][32][16];   // 16 KB
  __shared__ alignas(32) _Float16 Bsh[2][4][32][16];   //  8 KB

  const int t    = threadIdx.x;
  const int wv   = t >> 5;
  const int lane = t & 31;
  const int lm   = lane & 15;
  const int lh   = lane >> 4;
  const int m0   = blockIdx.x * 128;
  const int n0   = blockIdx.y * 64;

  // Decompose the two A-loader pixels owned by this thread (fixed over K).
  int  py[2], px[2];
  long pbase[2];
  #pragma unroll
  for (int i = 0; i < 2; ++i) {
    const int m  = m0 + (wv * 2 + i) * 16 + lm;
    const int ni = m >> hwShift;           // HoWo is pow2
    const int pr = m - (ni << hwShift);
    const int oy = pr / Wo;                // Wo pow2; strength-reduced
    const int ox = pr - oy * Wo;
    py[i]    = oy * stride - PAD;
    px[i]    = ox * stride - PAD;
    pbase[i] = ((long)ni * Hin * Win) << cinShift;
  }
  const _Float16* wrow = w16 + ((long)(n0 + (wv << 4) + lm)) * K;

  v8f acc[2][4] = {};

  // A-run address for (slot i, run g) at K-offset k0; valid=false -> pad zero
  auto aAddr = [&](int i, int g, int k0, bool& valid) -> const _Float16* {
    const int kb = k0 + g * 16 + (lh << 3);  // 8-run start, inside one ci-run
    const int rs = kb >> cinShift;
    const int cb = kb - (rs << cinShift);
    const int r  = rs / SS;
    const int s  = rs - r * SS;
    const int iy = py[i] + r;
    const int ix = px[i] + s;
    valid = (iy >= 0 && iy < Hin && ix >= 0 && ix < Win);
    return in + pbase[i] + (((long)(iy * Win + ix)) << cinShift) + cb;
  };

  // Stage K-tile k0 into LDS buffer `buf`.
  auto stageK = [&](int buf, int k0) {
#if USE_ASYNC_LDS
    const h8 z = {};
    #pragma unroll
    for (int i = 0; i < 2; ++i)
      #pragma unroll
      for (int g = 0; g < 2; ++g) {
        bool valid;
        const _Float16* gp = aAddr(i, g, k0, valid);
        _Float16* lp = &Ash[buf][wv * 2 + i][lane][g * 8];
        if (valid)
          ASYNC_CP16(gp, lp);
        else
          *(h8*)lp = z;
      }
    const _Float16* bp = wrow + k0 + (lh << 3);
    ASYNC_CP16(bp,      &Bsh[buf][wv][lane][0]);
    ASYNC_CP16(bp + 16, &Bsh[buf][wv][lane][8]);
#else
    h8 ra[2][2], rb[2];
    #pragma unroll
    for (int i = 0; i < 2; ++i)
      #pragma unroll
      for (int g = 0; g < 2; ++g) {
        bool valid;
        const _Float16* gp = aAddr(i, g, k0, valid);
        h8 v = {};
        if (valid) v = *(const h8*)gp;
        ra[i][g] = v;
      }
    const _Float16* bp = wrow + k0 + (lh << 3);
    rb[0] = *(const h8*)bp;
    rb[1] = *(const h8*)(bp + 16);
    #pragma unroll
    for (int i = 0; i < 2; ++i)
      #pragma unroll
      for (int g = 0; g < 2; ++g)
        *(h8*)(&Ash[buf][wv * 2 + i][lane][g * 8]) = ra[i][g];
    *(h8*)(&Bsh[buf][wv][lane][0]) = rb[0];
    *(h8*)(&Bsh[buf][wv][lane][8]) = rb[1];
#endif
  };

  auto computeK = [&](int buf) {
    v16h a0 = *(const v16h*)(&Ash[buf][wv * 2 + 0][lane][0]);
    v16h a1 = *(const v16h*)(&Ash[buf][wv * 2 + 1][lane][0]);
    #pragma unroll
    for (int nb = 0; nb < 4; ++nb) {
      v16h b = *(const v16h*)(&Bsh[buf][nb][lane][0]);
      acc[0][nb] = __builtin_amdgcn_wmma_f32_16x16x32_f16(
          false, a0, false, b, (short)0, acc[0][nb], false, false);
      acc[1][nb] = __builtin_amdgcn_wmma_f32_16x16x32_f16(
          false, a1, false, b, (short)0, acc[1][nb], false, false);
    }
  };

  // -------- double-buffered K pipeline ------------------------------------
  int p = 0;
  stageK(0, 0);
#if USE_ASYNC_LDS
  __builtin_amdgcn_s_wait_asynccnt(0);
#endif
  __syncthreads();
  for (int k0 = 32; k0 < K; k0 += 32) {
    stageK(p ^ 1, k0);     // issue next tile (async / reg-staged)
    computeK(p);           // overlap WMMAs with in-flight loads
#if USE_ASYNC_LDS
    __builtin_amdgcn_s_wait_asynccnt(0);
#endif
    __syncthreads();
    p ^= 1;
  }
  computeK(p);

  // ---- Epilogue: NHWC out index = pixel*Cout + c (no division) -----------
  #pragma unroll
  for (int nb = 0; nb < 4; ++nb) {
    const int   c     = n0 + nb * 16 + lm;
    const float scale = bng[c] * rsqrtf(bnrv[c] + BN_EPS);
    const float shift = bnb[c] - bnrm[c] * scale;
    #pragma unroll
    for (int i = 0; i < 2; ++i) {
      #pragma unroll
      for (int r = 0; r < 8; ++r) {
        const int  pix  = m0 + (wv * 2 + i) * 16 + (lh << 3) + r;
        const long oidx = (long)pix * Cout + c;
        float v = acc[i][nb][r] * scale + shift;
        if constexpr (EPI == 1) {
          const int ni = pix >> hwShift;
          v = fmaxf(v, 0.f) * mask[ni * Cout + c];
          ((_Float16*)outp)[oidx] = (_Float16)v;
        } else if constexpr (EPI == 2) {
          ((_Float16*)outp)[oidx] = (_Float16)v;
        } else {
          const int ni = pix >> hwShift;
          v = (v + (float)sc[oidx]) * mask[ni * Cout + c];
          ((float*)outp)[oidx] = fmaxf(v, 0.f);
        }
      }
    }
  }
}

// ---------------------------------------------------------------------------
// Stem conv: 3->64 ch, 3x3 s1 p1 on 256x3x32x32 (NCHW f32 input as given).
// K=27 (zero-padded to 32), single K-step; ~0.15% of total FLOPs.
// ---------------------------------------------------------------------------
__global__ __launch_bounds__(128) void stem_wmma_kernel(
    const float* __restrict__ in, const float* __restrict__ wgt,
    const float* __restrict__ bng, const float* __restrict__ bnb,
    const float* __restrict__ bnrm, const float* __restrict__ bnrv,
    float* __restrict__ outp)
{
  __shared__ alignas(32) _Float16 Asm[4][32][16];
  __shared__ alignas(32) _Float16 Bsm[4][32][16];
  const int t = threadIdx.x, frag = t >> 5, lane = t & 31;
  const int lm = lane & 15, lh = lane >> 4;
  const int m0 = blockIdx.x * 64;

  const int  mrow = m0 + frag * 16 + lm;
  const int  nimg = mrow >> 10;            // HoWo = 1024
  const int  prem = mrow & 1023;
  const int  oy = prem >> 5, ox = prem & 31;
  const int  iy0 = oy - 1, ix0 = ox - 1;
  const long inBase = (long)nimg * 3 * 32 * 32;
  const int  brow = frag * 16 + lm;

  #pragma unroll
  for (int j = 0; j < 16; ++j) {
    const int k = ((j >> 3) << 4) + (lh << 3) + (j & 7);
    _Float16 v = (_Float16)0.f;
    if (k < 27) {
      const int ci = k / 9, rsd = k - ci * 9, r = rsd / 3, s = rsd - r * 3;
      const int iy = iy0 + r, ix = ix0 + s;
      if (iy >= 0 && iy < 32 && ix >= 0 && ix < 32)
        v = (_Float16)in[inBase + (long)(ci * 32 + iy) * 32 + ix];
    }
    Asm[frag][lane][j] = v;
  }
  #pragma unroll
  for (int j = 0; j < 16; ++j) {
    const int k = ((j >> 3) << 4) + (lh << 3) + (j & 7);
    Bsm[frag][lane][j] = (k < 27) ? (_Float16)wgt[brow * 27 + k] : (_Float16)0.f;
  }
  __syncthreads();
  v8f acc[4] = {};
  v16h a = *(const v16h*)(&Asm[frag][lane][0]);
  #pragma unroll
  for (int nb = 0; nb < 4; ++nb) {
    v16h b = *(const v16h*)(&Bsm[nb][lane][0]);
    acc[nb] = __builtin_amdgcn_wmma_f32_16x16x32_f16(
        false, a, false, b, (short)0, acc[nb], false, false);
  }
  #pragma unroll
  for (int nb = 0; nb < 4; ++nb) {
    const int   c     = nb * 16 + lm;
    const float scale = bng[c] * rsqrtf(bnrv[c] + BN_EPS);
    const float shift = bnb[c] - bnrm[c] * scale;
    #pragma unroll
    for (int r = 0; r < 8; ++r) {
      const int p = m0 + frag * 16 + (lh << 3) + r;
      outp[(long)p * 64 + c] = fmaxf(acc[nb][r] * scale + shift, 0.f);
    }
  }
}

// ---- weight transform: f32 OIHW -> f16 [co][rs][ci]; co = blockIdx.y -----
__global__ void wtrans_kernel(const float* __restrict__ w, _Float16* __restrict__ o,
                              int cinShift, int RS, int K) {
  int k = blockIdx.x * 256 + threadIdx.x;
  if (k >= K) return;
  const int co = blockIdx.y;
  const int rs = k >> cinShift;
  const int ci = k & ((1 << cinShift) - 1);
  o[(long)co * K + k] = (_Float16)w[(((long)co << cinShift) + ci) * RS + rs];
}

// ---- per-stage soft channel mask: sigmoid(scores[ctx[b]][c]) / 0.25 ------
__global__ void stage_mask_kernel(const float* __restrict__ scores,
                                  const int* __restrict__ ctx,
                                  float* __restrict__ mask, int cShift, int n) {
  int i = blockIdx.x * blockDim.x + threadIdx.x;
  if (i >= n) return;
  int b = i >> cShift, c = i & ((1 << cShift) - 1);
  float s = scores[(ctx[b] << cShift) + c];
  mask[i] = (1.f / (1.f + expf(-s))) * 4.0f;
}

// ---- x_in = cur * mask(b,c), NHWC f32 -> NHWC f16 (mask may be null) -----
__global__ void mask_cvt_kernel(const float* __restrict__ in,
                                const float* __restrict__ mask,
                                _Float16* __restrict__ out,
                                int cShift, int hwcShift, long n) {
  long i = (long)blockIdx.x * blockDim.x + threadIdx.x;
  if (i >= n) return;
  int c = (int)(i & ((1 << cShift) - 1));
  int b = (int)(i >> hwcShift);
  float m = mask ? mask[(b << cShift) + c] : 1.f;
  out[i] = (_Float16)(in[i] * m);
}

// ---- global average pool over HW (NHWC) ----------------------------------
__global__ void pool_kernel(const float* __restrict__ in,
                            float* __restrict__ out, int cShift, int HW, int n) {
  int i = blockIdx.x * blockDim.x + threadIdx.x;
  if (i >= n) return;
  int b = i >> cShift, c = i & ((1 << cShift) - 1);
  const float* p = in + (((long)b * HW) << cShift) + c;
  float s = 0.f;
  for (int j = 0; j < HW; ++j) s += p[(long)j << cShift];
  out[i] = s / (float)HW;
}

// ---- final FC: logits[b][k] = feat[b] . fc_w[k] + fc_b[k] ----------------
__global__ void fc_kernel(const float* __restrict__ feat,
                          const float* __restrict__ w,
                          const float* __restrict__ b,
                          float* __restrict__ out, int C, int K, int n) {
  int i = blockIdx.x * blockDim.x + threadIdx.x;
  if (i >= n) return;
  int bi = i / K, k = i - bi * K;
  const float* f  = feat + (long)bi * C;
  const float* wr = w + (long)k * C;
  float s = b[k];
  for (int c = 0; c < C; ++c) s += f[c] * wr[c];
  out[i] = s;
}

static inline int ilog2(int v) { int s = 0; while ((1 << s) < v) ++s; return s; }

extern "C" void kernel_launch(void* const* d_in, const int* in_sizes, int n_in,
                              void* d_out, int out_size, void* d_ws, size_t ws_size,
                              hipStream_t stream) {
  (void)in_sizes; (void)n_in; (void)out_size; (void)ws_size;
  int idx = 0;
  auto nf = [&]() { return (const float*)d_in[idx++]; };
  const float* x   = nf();
  const int*   ctx = (const int*)d_in[idx++];
  const float* stem_w = nf();
  struct BN { const float *g, *b, *rm, *rv; };
  auto getBN = [&]() { BN p; p.g = nf(); p.b = nf(); p.rm = nf(); p.rv = nf(); return p; };
  BN stem_bn = getBN();

  const int SC_CH[4] = {64, 128, 256, 512};
  const int SC_NB[4] = {3, 4, 6, 3};
  struct Blk { const float* w1; BN bn1; const float* w2; BN bn2;
               const float* pw; BN pbn; int has_proj; int stride; int cin; int cout; };
  Blk blks[16];
  int nblk = 0, in_p = 64;
  for (int si = 0; si < 4; ++si) {
    int c = SC_CH[si];
    for (int bj = 0; bj < SC_NB[si]; ++bj) {
      int s = (bj == 0 && si > 0) ? 2 : 1;
      Blk bk{}; bk.stride = s; bk.cin = in_p; bk.cout = c;
      bk.w1 = nf(); bk.bn1 = getBN();
      bk.w2 = nf(); bk.bn2 = getBN();
      bk.has_proj = (s != 1 || in_p != c);
      if (bk.has_proj) { bk.pw = nf(); bk.pbn = getBN(); }
      blks[nblk++] = bk;
      in_p = c;
    }
  }
  const float* scores[4];
  for (int i = 0; i < 4; ++i) scores[i] = nf();
  const float* fc_w = nf();
  const float* fc_b = nf();

  // ---- workspace carve (deterministic) -----------------------------------
  char* ws = (char*)d_ws; size_t off = 0;
  auto carve = [&](size_t bytes) {
    void* p = ws + off; off += (bytes + 255) & ~(size_t)255; return p;
  };
  float* maskbuf[4];
  for (int i = 0; i < 4; ++i) maskbuf[i] = (float*)carve((size_t)256 * SC_CH[i] * 4);
  const size_t MAXF32 = (size_t)256 * 64 * 32 * 32 * sizeof(float);    // 64 MB
  float*    bufY0 = (float*)carve(MAXF32);              // NHWC f32 ping
  float*    bufY1 = (float*)carve(MAXF32);              // NHWC f32 pong
  _Float16* bufX  = (_Float16*)carve(MAXF32 / 2);       // masked input, f16 NHWC
  _Float16* bufH  = (_Float16*)carve(MAXF32 / 2);       // conv1 output, f16 NHWC
  _Float16* bufSC = (_Float16*)carve(MAXF32 / 2);       // shortcut,     f16 NHWC
  _Float16* wbuf  = (_Float16*)carve((size_t)512 * 512 * 9 * 2);  // 4.7 MB
  float*    feat  = (float*)carve((size_t)256 * 512 * 4);

  const int B = 256;

  // stage masks
  for (int si = 0; si < 4; ++si) {
    int n = B * SC_CH[si];
    stage_mask_kernel<<<(n + 255) / 256, 256, 0, stream>>>(
        scores[si], ctx, maskbuf[si], ilog2(SC_CH[si]), n);
  }
  // stem: NCHW f32 in -> NHWC f32 out (BN+ReLU)
  stem_wmma_kernel<<<dim3(B * 32 * 32 / 64, 1), 128, 0, stream>>>(
      x, stem_w, stem_bn.g, stem_bn.b, stem_bn.rm, stem_bn.rv, bufY0);

  float* cur = bufY0; float* nxt = bufY1;
  int H = 32, bi = 0;
  for (int si = 0; si < 4; ++si) {
    int c = SC_CH[si];
    const int coutShift = ilog2(c);
    for (int bj = 0; bj < SC_NB[si]; ++bj) {
      Blk& bk = blks[bi++];
      int s = bk.stride, Hin = H, Ho = H / s;
      const int cinShift = ilog2(bk.cin);
      const int hwShift  = ilog2(Ho * Ho);
      const float* m_in = (bj == 0) ? (si == 0 ? nullptr : maskbuf[si - 1]) : maskbuf[si];
      // x_in = cur * m_in  (NHWC f16)
      {
        long n = (long)B * Hin * Hin * bk.cin;
        mask_cvt_kernel<<<(unsigned)((n + 255) / 256), 256, 0, stream>>>(
            cur, m_in, bufX, cinShift, ilog2(Hin * Hin * bk.cin), n);
      }
      int M = B * Ho * Ho;
      dim3 grid(M / 128, c / 64);
      // conv1: transform weights, then 3x3 stride s, BN+ReLU*mask -> f16
      {
        int K = bk.cin * 9;
        wtrans_kernel<<<dim3((K + 255) / 256, c), 256, 0, stream>>>(bk.w1, wbuf, cinShift, 9, K);
        conv_wmma_nhwc<1, 3, 3><<<grid, 128, 0, stream>>>(
            bufX, wbuf, bk.bn1.g, bk.bn1.b, bk.bn1.rm, bk.bn1.rv,
            maskbuf[si], nullptr, bufH, cinShift, Hin, Hin, c, Ho, hwShift, s);
      }
      // shortcut
      const _Float16* scp;
      if (bk.has_proj) {
        int K = bk.cin;
        wtrans_kernel<<<dim3((K + 255) / 256, c), 256, 0, stream>>>(bk.pw, wbuf, cinShift, 1, K);
        conv_wmma_nhwc<2, 1, 1><<<grid, 128, 0, stream>>>(
            bufX, wbuf, bk.pbn.g, bk.pbn.b, bk.pbn.rm, bk.pbn.rv,
            nullptr, nullptr, bufSC, cinShift, Hin, Hin, c, Ho, hwShift, s);
        scp = bufSC;
      } else {
        scp = bufX;   // identity: dims match (s==1, cin==cout), already NHWC f16
      }
      // conv2: 3x3 s1, (BN + shortcut)*mask, ReLU -> f32 NHWC
      {
        int K = c * 9;
        wtrans_kernel<<<dim3((K + 255) / 256, c), 256, 0, stream>>>(bk.w2, wbuf, coutShift, 9, K);
        conv_wmma_nhwc<3, 3, 3><<<grid, 128, 0, stream>>>(
            bufH, wbuf, bk.bn2.g, bk.bn2.b, bk.bn2.rm, bk.bn2.rv,
            maskbuf[si], scp, nxt, coutShift, Ho, Ho, c, Ho, hwShift, 1);
      }
      float* tsw = cur; cur = nxt; nxt = tsw;
      H = Ho;
    }
  }
  // global average pool (NHWC, 4x4) -> feat (256,512)
  {
    int n = B * 512;
    pool_kernel<<<(n + 255) / 256, 256, 0, stream>>>(cur, feat, 9, H * H, n);
  }
  // FC -> logits (256,10)
  {
    int n = B * 10;
    fc_kernel<<<(n + 255) / 256, 256, 0, stream>>>(feat, fc_w, fc_b, (float*)d_out, 512, 10, n);
  }
}
